// BiDAF_39410619908591
// MI455X (gfx1250) — compile-verified, matrix operand order checked by reading, not run on hardware
//
#include <hip/hip_runtime.h>
#include <hip/hip_bf16.h>
#include <math.h>

typedef __attribute__((ext_vector_type(16))) __bf16 v16bf;
typedef __attribute__((ext_vector_type(8)))  float  v8f;
typedef __attribute__((ext_vector_type(4)))  unsigned int v4u;
typedef __attribute__((ext_vector_type(8)))  int v8i;
typedef __attribute__((ext_vector_type(4)))  int v4i;

#define TT 512
#define BB 32
#define JJ 64
#define D2 200
#define NG 300   // 3*H gates
#define HH 100
#define GDIM 800
#define SIMK 224 // 200 padded to 7 K-chunks of 32
#define DP 208   // 200 padded to 13 N-tiles of 16

__device__ __forceinline__ int koff(int r, int half) {
  return (r < 4) ? (2 * r + (half ? 8 : 0)) : (2 * (r - 4) + (half ? 24 : 16));
}

// Load a 16x32 bf16 A/B fragment from an LDS tile stored row-major
// (rows = M for A, rows = N for Bt), row stride `ld` (bf16 elements).
__device__ __forceinline__ v16bf load_frag(const __bf16* base, int ld) {
  int lane = threadIdx.x & 31;
  int half = lane >> 4;
  const __bf16* row = base + (size_t)(lane & 15) * ld;
  v16bf f;
#pragma unroll
  for (int r = 0; r < 8; ++r) {
    int k = koff(r, half);
    f[2 * r]     = row[k];
    f[2 * r + 1] = row[k + 1];
  }
  return f;
}

// Build a B fragment for Wh (global f32, [N][K] row-major), tile (nt,kc), with bounds/zero-pad.
__device__ __forceinline__ v16bf load_frag_wh(const float* Wh, int nt, int kc, int N, int K) {
  int lane = threadIdx.x & 31;
  int half = lane >> 4;
  int n = nt * 16 + (lane & 15);
  v16bf f;
#pragma unroll
  for (int r = 0; r < 8; ++r) {
    int k0 = kc * 32 + koff(r, half);
    float a = 0.f, b = 0.f;
    if (n < N) {
      if (k0 < K)     a = Wh[(size_t)n * K + k0];
      if (k0 + 1 < K) b = Wh[(size_t)n * K + k0 + 1];
    }
    f[2 * r]     = (__bf16)a;
    f[2 * r + 1] = (__bf16)b;
  }
  return f;
}

__device__ __forceinline__ float sigmoidf_(float x) { return 1.f / (1.f + __expf(-x)); }

// ---- Tensor Data Mover: DMA `rows` rows of `cols` f32 (row stride in elements)
// from global into LDS at byte offset lds_off. D# packing per CDNA5 ISA.
__device__ __forceinline__ void tdm_load_rows(const float* gbase, unsigned lds_off,
                                              unsigned rows, unsigned cols,
                                              unsigned row_stride_elems) {
  unsigned long long ga = (unsigned long long)(size_t)gbase;
  v4u g0;
  g0[0] = 1u;                                           // count=1 (user mode, gather off)
  g0[1] = lds_off;                                      // lds_addr (bytes)
  g0[2] = (unsigned)(ga & 0xFFFFFFFFu);                 // global_addr[31:0]
  g0[3] = (unsigned)((ga >> 32) & 0x01FFFFFFu) | (2u << 30); // global_addr[56:32] | type=2
  v8i g1;
  g1[0] = (int)(2u << 16);                              // wg_mask=0 | data_size=2 (4B) | no flags
  g1[1] = (int)((cols & 0xFFFFu) << 16);                // [47:32]=atomic_addr=0 | [63:48]=tensor_dim0 lo
  g1[2] = (int)(((cols >> 16) & 0xFFFFu) |              // [79:64]=tensor_dim0 hi
                ((rows & 0xFFFFu) << 16));              // [95:80]=tensor_dim1 lo
  g1[3] = (int)(((rows >> 16) & 0xFFFFu) |              // [111:96]=tensor_dim1 hi
                ((cols & 0xFFFFu) << 16));              // [127:112]=tile_dim0
  g1[4] = (int)(rows & 0xFFFFu);                        // [143:128]=tile_dim1 | tile_dim2=0
  g1[5] = (int)row_stride_elems;                        // tensor_dim0_stride[31:0]
  g1[6] = 0;                                            // stride0 hi | dim1_stride lo
  g1[7] = 0;
  v4i z4 = {0, 0, 0, 0};
#if __clang_major__ >= 23
  v8i z8 = {0, 0, 0, 0, 0, 0, 0, 0};
  __builtin_amdgcn_tensor_load_to_lds(g0, g1, z4, z4, z8, 0);
#else
  __builtin_amdgcn_tensor_load_to_lds(g0, g1, z4, z4, 0);
#endif
}

// ---------------- K0: ch[b,t] = c.w_h ; qu[b,j] = q.w_u ----------------
__global__ __launch_bounds__(256) void rowdot_kernel(const float* __restrict__ c,
                                                     const float* __restrict__ q,
                                                     const float* __restrict__ ws,
                                                     float* __restrict__ ch,
                                                     float* __restrict__ qu) {
  int row = blockIdx.x * 256 + threadIdx.x;
  if (row < BB * TT) {
    const float* p = c + (size_t)row * D2;
    float s = 0.f;
    for (int k = 0; k < D2; ++k) s += p[k] * ws[k];
    ch[row] = s;
  } else if (row < BB * TT + BB * JJ) {
    int r2 = row - BB * TT;
    const float* p = q + (size_t)r2 * D2;
    float s = 0.f;
    for (int k = 0; k < D2; ++k) s += p[k] * ws[D2 + k];
    qu[r2] = s;
  }
}

// ---------------- K1: sim = ch + qu + (c*w_hu) @ q^T  (WMMA) ----------------
__global__ __launch_bounds__(256) void sim_kernel(const float* __restrict__ c,
                                                  const float* __restrict__ q,
                                                  const float* __restrict__ ws,
                                                  const float* __restrict__ ch,
                                                  const float* __restrict__ qu,
                                                  float* __restrict__ sim) {
  __shared__ __bf16 Abuf[64 * SIMK];
  __shared__ __bf16 Bbuf[64 * SIMK];
  int b = blockIdx.y, t0 = blockIdx.x * 64;
  int tid = threadIdx.x;
  const float* whu = ws + 2 * D2;
  for (int i = tid; i < 64 * SIMK; i += 256) {
    int m = i / SIMK, k = i % SIMK;
    float av = 0.f, bv = 0.f;
    if (k < D2) {
      av = c[((size_t)b * TT + t0 + m) * D2 + k] * whu[k];
      bv = q[((size_t)b * JJ + m) * D2 + k];   // rows of Bt = j index (JJ==64)
    }
    Abuf[i] = (__bf16)av;
    Bbuf[i] = (__bf16)bv;
  }
  __syncthreads();
  int wave = tid >> 5, lane = tid & 31, half = lane >> 4;
#pragma unroll
  for (int i = 0; i < 2; ++i) {
    int ct = wave * 2 + i;
    int mt = ct >> 2, nt = ct & 3;
    v8f acc = {};
    for (int kc = 0; kc < 7; ++kc) {
      v16bf a  = load_frag(Abuf + (size_t)(mt * 16) * SIMK + kc * 32, SIMK);
      v16bf bb = load_frag(Bbuf + (size_t)(nt * 16) * SIMK + kc * 32, SIMK);
      acc = __builtin_amdgcn_wmma_f32_16x16x32_bf16(false, a, false, bb, (short)0, acc, false, false);
    }
    int n = nt * 16 + (lane & 15);
#pragma unroll
    for (int r = 0; r < 8; ++r) {
      int t = t0 + mt * 16 + (half ? 8 : 0) + r;
      sim[((size_t)b * TT + t) * JJ + n] = acc[r] + ch[b * TT + t] + qu[b * JJ + n];
    }
  }
}

// ---------------- K2a: rowmax over j ----------------
__global__ __launch_bounds__(256) void rowmax_kernel(const float* __restrict__ sim,
                                                     float* __restrict__ rowmax) {
  int row = blockIdx.x * 256 + threadIdx.x;
  if (row >= BB * TT) return;
  const float* p = sim + (size_t)row * JJ;
  float m = p[0];
  for (int j = 1; j < JJ; ++j) m = fmaxf(m, p[j]);
  rowmax[row] = m;
}

// ---------------- K2b: softmax over t for each (b,j), in place ----------------
__global__ __launch_bounds__(256) void colsoftmax_kernel(float* __restrict__ sim) {
  __shared__ float red[256];
  int b = blockIdx.y, j = blockIdx.x, tid = threadIdx.x;
  float v0 = sim[((size_t)b * TT + tid) * JJ + j];
  float v1 = sim[((size_t)b * TT + tid + 256) * JJ + j];
  red[tid] = fmaxf(v0, v1);
  __syncthreads();
  for (int s = 128; s > 0; s >>= 1) { if (tid < s) red[tid] = fmaxf(red[tid], red[tid + s]); __syncthreads(); }
  float mx = red[0];
  __syncthreads();
  float e0 = __expf(v0 - mx), e1 = __expf(v1 - mx);
  red[tid] = e0 + e1;
  __syncthreads();
  for (int s = 128; s > 0; s >>= 1) { if (tid < s) red[tid] += red[tid + s]; __syncthreads(); }
  float inv = 1.f / red[0];
  sim[((size_t)b * TT + tid) * JJ + j]       = e0 * inv;
  sim[((size_t)b * TT + tid + 256) * JJ + j] = e1 * inv;
}

// ---------------- K2c: attn_b = softmax_t(rowmax) ----------------
__global__ __launch_bounds__(256) void attnb_kernel(const float* __restrict__ rowmax,
                                                    float* __restrict__ attnb) {
  __shared__ float red[256];
  int b = blockIdx.x, tid = threadIdx.x;
  float v0 = rowmax[b * TT + tid];
  float v1 = rowmax[b * TT + tid + 256];
  red[tid] = fmaxf(v0, v1);
  __syncthreads();
  for (int s = 128; s > 0; s >>= 1) { if (tid < s) red[tid] = fmaxf(red[tid], red[tid + s]); __syncthreads(); }
  float mx = red[0];
  __syncthreads();
  float e0 = __expf(v0 - mx), e1 = __expf(v1 - mx);
  red[tid] = e0 + e1;
  __syncthreads();
  for (int s = 128; s > 0; s >>= 1) { if (tid < s) red[tid] += red[tid + s]; __syncthreads(); }
  float inv = 1.f / red[0];
  attnb[b * TT + tid]       = e0 * inv;
  attnb[b * TT + tid + 256] = e1 * inv;
}

// ---------------- K3: h_tilde[b,d] = sum_t attn_b[b,t]*c[b,t,d] ----------------
__global__ __launch_bounds__(256) void htilde_kernel(const float* __restrict__ attnb,
                                                     const float* __restrict__ c,
                                                     float* __restrict__ ht) {
  int b = blockIdx.x, d = threadIdx.x;
  if (d >= D2) return;
  float s = 0.f;
  for (int t = 0; t < TT; ++t) s += attnb[b * TT + t] * c[((size_t)b * TT + t) * D2 + d];
  ht[b * D2 + d] = s;
}

// ---------------- K4: u_tilde (WMMA) + assemble g ----------------
__global__ __launch_bounds__(256) void utilde_g_kernel(const float* __restrict__ attn,
                                                       const float* __restrict__ q,
                                                       const float* __restrict__ c,
                                                       const float* __restrict__ ht,
                                                       float* __restrict__ g) {
  __shared__ __bf16 Abuf[64 * 64];
  __shared__ __bf16 Bbuf[DP * 64];  // Bt[d][j]
  int b = blockIdx.y, t0 = blockIdx.x * 64;
  int tid = threadIdx.x;
  for (int i = tid; i < 64 * 64; i += 256) {
    int m = i >> 6, k = i & 63;
    Abuf[i] = (__bf16)attn[((size_t)b * TT + t0 + m) * JJ + k];
  }
  for (int i = tid; i < DP * 64; i += 256) {
    int d = i >> 6, j = i & 63;
    float v = (d < D2) ? q[((size_t)b * JJ + j) * D2 + d] : 0.f;  // transposed stage
    Bbuf[i] = (__bf16)v;
  }
  __syncthreads();
  int wave = tid >> 5, lane = tid & 31, half = lane >> 4;
  for (int ct = wave; ct < 4 * 13; ct += 8) {
    int mt = ct / 13, nt = ct % 13;
    v8f acc = {};
#pragma unroll
    for (int kc = 0; kc < 2; ++kc) {
      v16bf a  = load_frag(Abuf + (size_t)(mt * 16) * 64 + kc * 32, 64);
      v16bf bb = load_frag(Bbuf + (size_t)(nt * 16) * 64 + kc * 32, 64);
      acc = __builtin_amdgcn_wmma_f32_16x16x32_bf16(false, a, false, bb, (short)0, acc, false, false);
    }
    int d = nt * 16 + (lane & 15);
    if (d < D2) {
#pragma unroll
      for (int r = 0; r < 8; ++r) {
        int t = t0 + mt * 16 + (half ? 8 : 0) + r;
        float u  = acc[r];
        float cv = c[((size_t)b * TT + t) * D2 + d];
        size_t gb = ((size_t)b * TT + t) * GDIM;
        g[gb + d]            = cv;
        g[gb + D2 + d]       = u;
        g[gb + 2 * D2 + d]   = cv * u;
        g[gb + 3 * D2 + d]   = cv * ht[b * D2 + d];
      }
    }
  }
}

// ---------------- K6: Y[16384,300] = X[16384,K] @ W[N,K]^T + bias (WMMA) ----------------
__global__ __launch_bounds__(256) void proj_kernel(const float* __restrict__ X,
                                                   const float* __restrict__ W,
                                                   const float* __restrict__ bias,
                                                   float* __restrict__ Y, int K) {
  __shared__ __bf16 Abuf[64 * 32];
  __shared__ __bf16 Bbuf[64 * 32];
  int r0 = blockIdx.x * 64, n0 = blockIdx.y * 64;
  int tid = threadIdx.x, wave = tid >> 5, lane = tid & 31, half = lane >> 4;
  int nk = (K + 31) / 32;
  int ct0 = wave * 2, ct1 = wave * 2 + 1;
  int mt0 = ct0 >> 2, nt0 = ct0 & 3, mt1 = ct1 >> 2, nt1 = ct1 & 3;
  v8f acc0 = {}, acc1 = {};
  for (int kc = 0; kc < nk; ++kc) {
    for (int i = tid; i < 64 * 32; i += 256) {
      int m = i >> 5, k = i & 31;
      int kg = kc * 32 + k;
      float av = (kg < K) ? X[(size_t)(r0 + m) * K + kg] : 0.f;
      int n = n0 + m;
      float bv = (kg < K && n < NG) ? W[(size_t)n * K + kg] : 0.f;
      Abuf[i] = (__bf16)av;
      Bbuf[i] = (__bf16)bv;
    }
    __syncthreads();
    {
      v16bf a = load_frag(Abuf + (size_t)(mt0 * 16) * 32, 32);
      v16bf b = load_frag(Bbuf + (size_t)(nt0 * 16) * 32, 32);
      acc0 = __builtin_amdgcn_wmma_f32_16x16x32_bf16(false, a, false, b, (short)0, acc0, false, false);
      a = load_frag(Abuf + (size_t)(mt1 * 16) * 32, 32);
      b = load_frag(Bbuf + (size_t)(nt1 * 16) * 32, 32);
      acc1 = __builtin_amdgcn_wmma_f32_16x16x32_bf16(false, a, false, b, (short)0, acc1, false, false);
    }
    __syncthreads();
  }
  int n_0 = n0 + nt0 * 16 + (lane & 15);
  int n_1 = n0 + nt1 * 16 + (lane & 15);
#pragma unroll
  for (int r = 0; r < 8; ++r) {
    int row0 = r0 + mt0 * 16 + (half ? 8 : 0) + r;
    int row1 = r0 + mt1 * 16 + (half ? 8 : 0) + r;
    if (n_0 < NG) Y[(size_t)row0 * NG + n_0] = acc0[r] + bias[n_0];
    if (n_1 < NG) Y[(size_t)row1 * NG + n_1] = acc1[r] + bias[n_1];
  }
}

// ---------------- K7: GRU scan: WMMA recurrence + TDM double-buffered gx prefetch ----------------
__global__ __launch_bounds__(320) void gru_scan_kernel(const float* __restrict__ gxF,
                                                       const float* __restrict__ gxB,
                                                       const float* __restrict__ Whh,
                                                       const float* __restrict__ bhh,
                                                       float* __restrict__ out) {
  int dir = blockIdx.x;
  const float* gx = dir ? gxB : gxF;
  const float* Wh = Whh + (size_t)dir * NG * HH;
  const float* bh = bhh + dir * NG;
  __shared__ __bf16 hbf[BB * 128];         // h as bf16, K padded to 128
  __shared__ float  hf[BB * HH];           // h fp32 state
  __shared__ float  gh[BB * NG];           // h @ Wh^T + bhh
  __shared__ float  gxbuf[2][BB * NG];     // TDM double buffer for this step's gx rows
  __shared__ float  bhl[320];
  int tid = threadIdx.x, wave = tid >> 5, lane = tid & 31, half = lane >> 4;
  for (int i = tid; i < BB * 128; i += 320) hbf[i] = (__bf16)0.f;
  for (int i = tid; i < BB * HH; i += 320) hf[i] = 0.f;
  bhl[tid] = (tid < NG) ? bh[tid] : 0.f;
  // Pin this wave's Wh B-fragments (n-tiles wave*2, wave*2+1) in registers.
  v16bf bw[2][4];
#pragma unroll
  for (int i = 0; i < 2; ++i)
#pragma unroll
    for (int kc = 0; kc < 4; ++kc)
      bw[i][kc] = load_frag_wh(Wh, wave * 2 + i, kc, NG, HH);
  // Kick off TDM for the first step's gx rows (32 rows, stride TT*NG elements).
  if (wave == 0) {
    int t0i = dir ? (TT - 1) : 0;
    tdm_load_rows(gx + (size_t)t0i * NG, (unsigned)(size_t)(void*)&gxbuf[0][0],
                  BB, NG, (unsigned)(TT * NG));
  }
  __syncthreads();
  for (int tt = 0; tt < TT; ++tt) {
    int t = dir ? (TT - 1 - tt) : tt;
    // gh = h @ Wh^T + bhh : 2 M-tiles x 2 N-tiles per wave, K = 4 chunks.
    // Overlaps with the in-flight TDM prefetch of this step's gx.
#pragma unroll
    for (int mt = 0; mt < 2; ++mt) {
#pragma unroll
      for (int i = 0; i < 2; ++i) {
        v8f acc = {};
#pragma unroll
        for (int kc = 0; kc < 4; ++kc) {
          v16bf a = load_frag(hbf + (size_t)(mt * 16) * 128 + kc * 32, 128);
          acc = __builtin_amdgcn_wmma_f32_16x16x32_bf16(false, a, false, bw[i][kc], (short)0, acc, false, false);
        }
        int n = (wave * 2 + i) * 16 + (lane & 15);
        if (n < NG) {
#pragma unroll
          for (int r = 0; r < 8; ++r) {
            int bb = mt * 16 + (half ? 8 : 0) + r;
            gh[bb * NG + n] = acc[r] + bhl[n];
          }
        }
      }
    }
    if (wave == 0) {
      // Ensure this step's gx tile has landed, then prefetch the next step's
      // tile into the other buffer (its previous readers finished last step).
      __builtin_amdgcn_s_wait_tensorcnt(0);
      if (tt + 1 < TT) {
        int tn = dir ? (TT - 2 - tt) : (tt + 1);
        tdm_load_rows(gx + (size_t)tn * NG, (unsigned)(size_t)(void*)&gxbuf[(tt + 1) & 1][0],
                      BB, NG, (unsigned)(TT * NG));
      }
    }
    __syncthreads();
    // gate update + state write (reads gx from LDS, not global)
    const float* gcur = &gxbuf[tt & 1][0];
    for (int idx = tid; idx < BB * HH; idx += 320) {
      int bb = idx / HH, hh = idx % HH;
      const float* gr = gcur + bb * NG;
      float rg = sigmoidf_(gr[hh] + gh[bb * NG + hh]);
      float zg = sigmoidf_(gr[HH + hh] + gh[bb * NG + HH + hh]);
      float ng = tanhf(gr[2 * HH + hh] + rg * gh[bb * NG + 2 * HH + hh]);
      float hn = (1.f - zg) * ng + zg * hf[idx];
      hf[idx] = hn;
      hbf[bb * 128 + hh] = (__bf16)hn;
      out[((size_t)bb * TT + t) * (2 * HH) + dir * HH + hh] = hn;
    }
    __syncthreads();
  }
}

// ---------------- K8: span scores ----------------
__global__ __launch_bounds__(256) void scores_kernel(const float* __restrict__ g,
                                                     const float* __restrict__ mS,
                                                     const float* __restrict__ mE,
                                                     const float* __restrict__ wps,
                                                     const float* __restrict__ wpe,
                                                     float* __restrict__ sS,
                                                     float* __restrict__ sE) {
  int row = blockIdx.x * 256 + threadIdx.x;
  if (row >= BB * TT) return;
  const float* gr = g + (size_t)row * GDIM;
  const float* ms = mS + (size_t)row * (2 * HH);
  const float* me = mE + (size_t)row * (2 * HH);
  float a = 0.f, b = 0.f;
  for (int i = 0; i < GDIM; ++i) { float gv = gr[i]; a += gv * wps[i]; b += gv * wpe[i]; }
  for (int i = 0; i < 2 * HH; ++i) { a += ms[i] * wps[GDIM + i]; b += me[i] * wpe[GDIM + i]; }
  sS[row] = a;
  sE[row] = b;
}

// ---------------- K9: final softmax over t ----------------
__global__ __launch_bounds__(256) void psoftmax_kernel(const float* __restrict__ sS,
                                                       const float* __restrict__ sE,
                                                       float* __restrict__ out) {
  __shared__ float red[256];
  int b = blockIdx.x, which = blockIdx.y, tid = threadIdx.x;
  const float* s = which ? sE : sS;
  float* o = out + (size_t)which * BB * TT;
  float v0 = s[b * TT + tid];
  float v1 = s[b * TT + tid + 256];
  red[tid] = fmaxf(v0, v1);
  __syncthreads();
  for (int st = 128; st > 0; st >>= 1) { if (tid < st) red[tid] = fmaxf(red[tid], red[tid + st]); __syncthreads(); }
  float mx = red[0];
  __syncthreads();
  float e0 = __expf(v0 - mx), e1 = __expf(v1 - mx);
  red[tid] = e0 + e1;
  __syncthreads();
  for (int st = 128; st > 0; st >>= 1) { if (tid < st) red[tid] += red[tid + st]; __syncthreads(); }
  float inv = 1.f / red[0];
  o[b * TT + tid]       = e0 * inv;
  o[b * TT + tid + 256] = e1 * inv;
}

extern "C" void kernel_launch(void* const* d_in, const int* in_sizes, int n_in,
                              void* d_out, int out_size, void* d_ws, size_t ws_size,
                              hipStream_t stream) {
  const float* c        = (const float*)d_in[0];
  const float* q        = (const float*)d_in[1];
  const float* w_s      = (const float*)d_in[2];
  const float* mod_Wih0 = (const float*)d_in[3];
  const float* mod_Whh0 = (const float*)d_in[4];
  const float* mod_bih0 = (const float*)d_in[5];
  const float* mod_bhh0 = (const float*)d_in[6];
  const float* mod_Wih1 = (const float*)d_in[7];
  const float* mod_Whh1 = (const float*)d_in[8];
  const float* mod_bih1 = (const float*)d_in[9];
  const float* mod_bhh1 = (const float*)d_in[10];
  const float* out_Wih  = (const float*)d_in[11];
  const float* out_Whh  = (const float*)d_in[12];
  const float* out_bih  = (const float*)d_in[13];
  const float* out_bhh  = (const float*)d_in[14];
  const float* wps      = (const float*)d_in[15];
  const float* wpe      = (const float*)d_in[16];

  float* w = (float*)d_ws;
  size_t off = 0;
  float* ch     = w + off; off += (size_t)BB * TT;
  float* qu     = w + off; off += (size_t)BB * JJ;
  float* rowmax = w + off; off += (size_t)BB * TT;
  float* attnb  = w + off; off += (size_t)BB * TT;
  float* htl    = w + off; off += (size_t)BB * D2;
  float* sS     = w + off; off += (size_t)BB * TT;
  float* sE     = w + off; off += (size_t)BB * TT;
  float* sim    = w + off; off += (size_t)BB * TT * JJ;
  float* g      = w + off; off += (size_t)BB * TT * GDIM;
  float* gxF    = w + off; off += (size_t)BB * TT * NG;
  float* gxB    = w + off; off += (size_t)BB * TT * NG;
  float* mA     = w + off; off += (size_t)BB * TT * 2 * HH;
  float* mB     = w + off; off += (size_t)BB * TT * 2 * HH;
  float* mC     = w + off; off += (size_t)BB * TT * 2 * HH;

  // ---- attention ----
  rowdot_kernel<<<(BB * TT + BB * JJ + 255) / 256, 256, 0, stream>>>(c, q, w_s, ch, qu);
  sim_kernel<<<dim3(TT / 64, BB), 256, 0, stream>>>(c, q, w_s, ch, qu, sim);
  rowmax_kernel<<<(BB * TT) / 256, 256, 0, stream>>>(sim, rowmax);
  colsoftmax_kernel<<<dim3(JJ, BB), 256, 0, stream>>>(sim);   // sim -> attn_a in place
  attnb_kernel<<<BB, 256, 0, stream>>>(rowmax, attnb);
  htilde_kernel<<<BB, 256, 0, stream>>>(attnb, c, htl);
  utilde_g_kernel<<<dim3(TT / 64, BB), 256, 0, stream>>>(sim, q, c, htl, g);

  // ---- BiGRU layer 0 (input = g, K=800) ----
  proj_kernel<<<dim3(BB * TT / 64, 5), 256, 0, stream>>>(g, mod_Wih0, mod_bih0, gxF, GDIM);
  proj_kernel<<<dim3(BB * TT / 64, 5), 256, 0, stream>>>(g, mod_Wih0 + (size_t)NG * GDIM, mod_bih0 + NG, gxB, GDIM);
  gru_scan_kernel<<<2, 320, 0, stream>>>(gxF, gxB, mod_Whh0, mod_bhh0, mA);

  // ---- BiGRU layer 1 (input = mA, K=200) ----
  proj_kernel<<<dim3(BB * TT / 64, 5), 256, 0, stream>>>(mA, mod_Wih1, mod_bih1, gxF, 2 * HH);
  proj_kernel<<<dim3(BB * TT / 64, 5), 256, 0, stream>>>(mA, mod_Wih1 + (size_t)NG * 2 * HH, mod_bih1 + NG, gxB, 2 * HH);
  gru_scan_kernel<<<2, 320, 0, stream>>>(gxF, gxB, mod_Whh1, mod_bhh1, mB);

  // ---- output BiGRU (input = mB, K=200) ----
  proj_kernel<<<dim3(BB * TT / 64, 5), 256, 0, stream>>>(mB, out_Wih, out_bih, gxF, 2 * HH);
  proj_kernel<<<dim3(BB * TT / 64, 5), 256, 0, stream>>>(mB, out_Wih + (size_t)NG * 2 * HH, out_bih + NG, gxB, 2 * HH);
  gru_scan_kernel<<<2, 320, 0, stream>>>(gxF, gxB, out_Whh, out_bhh, mC);

  // ---- span heads ----
  scores_kernel<<<(BB * TT) / 256, 256, 0, stream>>>(g, mB, mC, wps, wpe, sS, sE);
  psoftmax_kernel<<<dim3(BB, 2), 256, 0, stream>>>(sS, sE, (float*)d_out);
}
